// MultiHeadSelfAttention_67508295959207
// MI455X (gfx1250) — matrix operand order repaired and checked by reading, not compiled, not run on hardware
//
#include <hip/hip_runtime.h>
#include <hip/hip_bf16.h>

#define B_  8
#define N_  2048
#define D_  384
#define H_  6
#define DH_ 64
#define BN_ (B_*N_)
#define D3_ (3*D_)

static constexpr float ATTN_SCALE = 0.125f;   // 64^-0.5

typedef __bf16 bf16;
typedef __attribute__((ext_vector_type(16))) __bf16 v16bf;
typedef __attribute__((ext_vector_type(8)))  __bf16 v8bf;
typedef __attribute__((ext_vector_type(8)))  float  v8f;
typedef __attribute__((ext_vector_type(4)))  unsigned s4u;   // SGPR quad (D# grp0/2/3)
typedef __attribute__((ext_vector_type(8)))  unsigned s8u;   // SGPR octet (D# grp1)

__device__ __forceinline__ v16bf cat8(v8bf a, v8bf b) {
    return __builtin_shufflevector(a, b, 0,1,2,3,4,5,6,7,8,9,10,11,12,13,14,15);
}
__device__ __forceinline__ v8bf ld8(const bf16* p) {
    return *reinterpret_cast<const v8bf*>(p);
}
__device__ __forceinline__ v8f wmma_bf16(v16bf a, v16bf b, v8f c) {
    // D = A(16x32 bf16) * B(32x16 bf16) + C(16x16 f32)
    return __builtin_amdgcn_wmma_f32_16x16x32_bf16(false, a, false, b,
                                                   (short)0, c, false, false);
}
__device__ __forceinline__ float rowmax16(float v) {
    v = fmaxf(v, __shfl_xor(v, 1, 16));
    v = fmaxf(v, __shfl_xor(v, 2, 16));
    v = fmaxf(v, __shfl_xor(v, 4, 16));
    v = fmaxf(v, __shfl_xor(v, 8, 16));
    return v;
}
__device__ __forceinline__ float rowsum16(float v) {
    v += __shfl_xor(v, 1, 16);
    v += __shfl_xor(v, 2, 16);
    v += __shfl_xor(v, 4, 16);
    v += __shfl_xor(v, 8, 16);
    return v;
}

// TDM: DMA a [rows x 384] bf16 panel (row-major, stride 384) into LDS.
// Issued by the calling wave; EXEC-independent; tracked by TENSORcnt.
__device__ __forceinline__ void tdm_load_panel(const bf16* gsrc, unsigned lds_off,
                                               unsigned rows) {
    const unsigned long long ga = (unsigned long long)(uintptr_t)gsrc;
    s4u g0; s8u g1; s4u gz = {0, 0, 0, 0};
    g0[0] = 1u;                                           // count=1, user mode
    g0[1] = lds_off;                                      // lds_addr
    g0[2] = (unsigned)ga;                                 // global_addr[31:0]
    g0[3] = ((unsigned)(ga >> 32) & 0x01FFFFFFu)          // global_addr[56:32]
            | 0x80000000u;                                // type=2 ("image")
    g1[0] = 1u << 16;                                     // data_size=2B, no mask
    g1[1] = (unsigned)D_ << 16;                           // tensor_dim0[15:0]
    g1[2] = (rows & 0xFFFFu) << 16;                       // dim0 hi=0 | tensor_dim1 lo
    g1[3] = (unsigned)D_ << 16;                           // dim1 hi=0 | tile_dim0=384
    g1[4] = rows & 0xFFFFu;                               // tile_dim1=rows, tile_dim2=0
    g1[5] = (unsigned)D_;                                 // tensor_dim0_stride lo32
    g1[6] = 0u;                                           // stride hi | dim1_stride lo
    g1[7] = 0u;
    asm volatile("tensor_load_to_lds %0, %1, %2, %3"
                 :: "s"(g0), "s"(g1), "s"(gz), "s"(gz) : "memory");
    asm volatile("s_wait_tensorcnt 0" ::: "memory");
}

// ---------------------------------------------------------------------------
// One-shot fp32 -> bf16 conversion (x, w_qkv, w_out). 8 elems / thread.
// ---------------------------------------------------------------------------
__global__ __launch_bounds__(256) void cvt_kernel(
    const float* __restrict__ in, bf16* __restrict__ out, int n8)
{
    const int i = blockIdx.x * blockDim.x + threadIdx.x;
    if (i < n8) {
        const float4* p = reinterpret_cast<const float4*>(in) + (size_t)i * 2;
        const float4 a = p[0], b = p[1];
        v8bf o;
        o[0] = (bf16)a.x; o[1] = (bf16)a.y; o[2] = (bf16)a.z; o[3] = (bf16)a.w;
        o[4] = (bf16)b.x; o[5] = (bf16)b.y; o[6] = (bf16)b.z; o[7] = (bf16)b.w;
        reinterpret_cast<v8bf*>(out)[i] = o;
    }
}

// ---------------------------------------------------------------------------
// Kernel 1: QKV projection. One block = one 64-col weight panel (TDM -> LDS,
// shared by 8 waves) x 128 rows. Each wave: 16x64 tile, 4 wmma per A load.
// Q pre-scaled; V stored transposed [B,H,Dh,N].
// ---------------------------------------------------------------------------
__global__ __launch_bounds__(256) void qkv_proj_kernel(
    const bf16* __restrict__ xbf, const bf16* __restrict__ wbf,
    const float* __restrict__ bias,
    bf16* __restrict__ qws, bf16* __restrict__ kws, bf16* __restrict__ vtws)
{
    __shared__ __align__(16) bf16 wlds[64 * D_];         // 48KB weight panel

    const int NT64 = D3_ / 64;                           // 18 column panels
    const int wave = threadIdx.x >> 5;
    const int lane = threadIdx.x & 31;
    const int hi = lane >> 4, l16 = lane & 15;
    const int nt   = blockIdx.x % NT64;                  // block-uniform panel
    const int mblk = blockIdx.x / NT64;

    if (wave == 0)
        tdm_load_panel(wbf + (size_t)nt * 64 * D_,
                       (unsigned)(uintptr_t)(&wlds[0]), 64);
    __syncthreads();

    const bf16* xrow = xbf + (size_t)(mblk * 128 + wave * 16 + l16) * D_;
    const bf16* l0 = wlds + (size_t)l16 * D_;            // B cols from LDS panel
    const bf16* l1 = l0 + (size_t)16 * D_;
    const bf16* l2 = l0 + (size_t)32 * D_;
    const bf16* l3 = l0 + (size_t)48 * D_;

    v8f acc[4] = {{}, {}, {}, {}};
    for (int k = 0; k < D_; k += 32) {
        const int ca = k + (hi ? 16 : 0);                // A K-chunks
        const int kb = k + (hi ? 16 : 0);                // B K rows
        const v16bf A = cat8(ld8(xrow + ca), ld8(xrow + ca + 8));
        acc[0] = wmma_bf16(A, cat8(ld8(l0 + kb), ld8(l0 + kb + 8)), acc[0]);
        acc[1] = wmma_bf16(A, cat8(ld8(l1 + kb), ld8(l1 + kb + 8)), acc[1]);
        acc[2] = wmma_bf16(A, cat8(ld8(l2 + kb), ld8(l2 + kb + 8)), acc[2]);
        acc[3] = wmma_bf16(A, cat8(ld8(l3 + kb), ld8(l3 + kb + 8)), acc[3]);
    }

    const int ebase = nt * 64;          // 64-wide panel never straddles Q/K/V or heads
    const int which = ebase / D_;       // 0=Q 1=K 2=V (uniform per block)
    const int h     = (ebase - which * D_) >> 6;

    #pragma unroll
    for (int sub = 0; sub < 4; ++sub) {
        const int e  = ebase + sub * 16 + l16;
        const int dd = sub * 16 + l16;
        const float bv = bias[e];
        #pragma unroll
        for (int r = 0; r < 8; ++r) {
            const int grow = mblk * 128 + wave * 16 + r + (hi ? 8 : 0);
            const int b = grow >> 11, n = grow & 2047;
            const float val = acc[sub][r] + bv;
            const size_t bh = (size_t)b * H_ + h;
            if (which == 0)      qws[(bh * N_ + n) * DH_ + dd] = (bf16)(val * ATTN_SCALE);
            else if (which == 1) kws[(bh * N_ + n) * DH_ + dd] = (bf16)val;
            else                 vtws[(bh * DH_ + dd) * N_ + n] = (bf16)val;
        }
    }
}

// ---------------------------------------------------------------------------
// Kernel 2: flash attention. 8 waves/block, wave = 16 query rows.
// K (32x64) and V^T (64x32) tiles double-buffered in LDS via
// global_load_async_to_lds_b128 (ASYNCcnt), shared by all 8 waves.
// ---------------------------------------------------------------------------
__global__ __launch_bounds__(256) void flash_attn_kernel(
    const bf16* __restrict__ qws, const bf16* __restrict__ kws,
    const bf16* __restrict__ vtws, bf16* __restrict__ attn)
{
    __shared__ __align__(16) bf16 kt[2][32][64];    // 8KB: 32 keys x 64 dims
    __shared__ __align__(16) bf16 vts[2][64][32];   // 8KB: 64 dims x 32 keys
    __shared__ __align__(16) bf16 pb[8][16][32];    // 8KB: per-wave P tile

    const int tid  = threadIdx.x;
    const int wave = tid >> 5, lane = tid & 31;
    const int hi = lane >> 4, l16 = lane & 15;

    const int blocksPerBH = N_ / 128;               // 16
    const int bh   = blockIdx.x / blocksPerBH;
    const int qblk = blockIdx.x % blocksPerBH;
    const int b = bh / H_, h = bh % H_;
    const int q0 = qblk * 128 + wave * 16;
    const int c0 = hi ? 8 : 0, d0 = hi ? 16 : 0;

    const bf16* kbh = kws  + (size_t)bh * N_ * DH_;
    const bf16* vbh = vtws + (size_t)bh * DH_ * N_;
    const int vdim = tid >> 2, vch = tid & 3;       // V^T staging split

    auto stage = [&](int jj, int s) {
        unsigned lk = (unsigned)(uintptr_t)(&kt[s][0][0] + (size_t)tid * 8);
        unsigned long long gk =
            (unsigned long long)(uintptr_t)(kbh + (size_t)jj * DH_ + (size_t)tid * 8);
        asm volatile("global_load_async_to_lds_b128 %0, %1, off"
                     :: "v"(lk), "v"(gk) : "memory");
        unsigned lv = (unsigned)(uintptr_t)(&vts[s][vdim][vch * 8]);
        unsigned long long gv =
            (unsigned long long)(uintptr_t)(vbh + (size_t)vdim * N_ + jj + vch * 8);
        asm volatile("global_load_async_to_lds_b128 %0, %1, off"
                     :: "v"(lv), "v"(gv) : "memory");
    };

    const bf16* qrow = qws + ((size_t)bh * N_ + q0 + l16) * DH_;
    const v16bf aq0 = cat8(ld8(qrow + d0),      ld8(qrow + d0 + 8));
    const v16bf aq1 = cat8(ld8(qrow + 32 + d0), ld8(qrow + 32 + d0 + 8));

    v8f o0 = {}, o1 = {}, o2 = {}, o3 = {};
    float mrow[8], lrow[8];
    #pragma unroll
    for (int r = 0; r < 8; ++r) { mrow[r] = -1e30f; lrow[r] = 0.f; }

    stage(0, 0);
    const int ITERS = N_ / 32;                      // 64
    for (int i = 0; i < ITERS; ++i) {
        const int s = i & 1;
        asm volatile("s_wait_asynccnt 0" ::: "memory");
        __syncthreads();                            // tile s fully in LDS
        if (i + 1 < ITERS) stage((i + 1) * 32, s ^ 1);

        // --- S = Q K^T for 32 keys (K B-fragments from LDS) ---
        const bf16* kr0 = &kt[s][l16][d0];
        const bf16* kr1 = &kt[s][16 + l16][d0];
        v8f s0 = {}, s1 = {};
        s0 = wmma_bf16(aq0, cat8(ld8(kr0),      ld8(kr0 + 8)),  s0);
        s0 = wmma_bf16(aq1, cat8(ld8(kr0 + 32), ld8(kr0 + 40)), s0);
        s1 = wmma_bf16(aq0, cat8(ld8(kr1),      ld8(kr1 + 8)),  s1);
        s1 = wmma_bf16(aq1, cat8(ld8(kr1 + 32), ld8(kr1 + 40)), s1);

        // --- online softmax; rows live per-VGPR; 16-lane reductions ---
        #pragma unroll
        for (int r = 0; r < 8; ++r) {
            float mx   = rowmax16(fmaxf(s0[r], s1[r]));
            float mnew = fmaxf(mrow[r], mx);
            float corr = __expf(mrow[r] - mnew);
            mrow[r] = mnew;
            float e0 = __expf(s0[r] - mnew);
            float e1 = __expf(s1[r] - mnew);
            lrow[r] = lrow[r] * corr + rowsum16(e0 + e1);
            o0[r] *= corr; o1[r] *= corr; o2[r] *= corr; o3[r] *= corr;
            pb[wave][r + (hi ? 8 : 0)][l16]      = (bf16)e0;   // C-layout -> LDS
            pb[wave][r + (hi ? 8 : 0)][16 + l16] = (bf16)e1;
        }
        asm volatile("s_wait_dscnt 0" ::: "memory");           // same-wave LDS RAW

        // --- reload P as A-fragment; O += P * V (V B-fragments from LDS) ---
        const bf16* prow = &pb[wave][l16][0];
        const v16bf ap = cat8(ld8(prow + d0), ld8(prow + d0 + 8));
        const bf16* v0 = &vts[s][l16][d0];
        const bf16* v1 = &vts[s][16 + l16][d0];
        const bf16* v2 = &vts[s][32 + l16][d0];
        const bf16* v3 = &vts[s][48 + l16][d0];
        o0 = wmma_bf16(ap, cat8(ld8(v0), ld8(v0 + 8)), o0);
        o1 = wmma_bf16(ap, cat8(ld8(v1), ld8(v1 + 8)), o1);
        o2 = wmma_bf16(ap, cat8(ld8(v2), ld8(v2 + 8)), o2);
        o3 = wmma_bf16(ap, cat8(ld8(v3), ld8(v3 + 8)), o3);
    }

    // --- normalize and write [B,N,H*Dh] bf16 for the output GEMM ---
    #pragma unroll
    for (int r = 0; r < 8; ++r) {
        const float inv = 1.0f / lrow[r];
        const int n = q0 + r + (hi ? 8 : 0);
        bf16* arow = attn + ((size_t)b * N_ + n) * D_ + h * DH_;
        arow[l16]      = (bf16)(o0[r] * inv);
        arow[16 + l16] = (bf16)(o1[r] * inv);
        arow[32 + l16] = (bf16)(o2[r] * inv);
        arow[48 + l16] = (bf16)(o3[r] * inv);
    }
}

// ---------------------------------------------------------------------------
// Kernel 3: output projection. Weight panel via TDM -> LDS; fp32 out + bias.
// ---------------------------------------------------------------------------
__global__ __launch_bounds__(256) void out_proj_kernel(
    const bf16* __restrict__ attn, const bf16* __restrict__ wbf,
    const float* __restrict__ bias, float* __restrict__ out)
{
    __shared__ __align__(16) bf16 wlds[64 * D_];         // 48KB weight panel

    const int NT64 = D_ / 64;                            // 6 column panels
    const int wave = threadIdx.x >> 5;
    const int lane = threadIdx.x & 31;
    const int hi = lane >> 4, l16 = lane & 15;
    const int nt   = blockIdx.x % NT64;
    const int mblk = blockIdx.x / NT64;

    if (wave == 0)
        tdm_load_panel(wbf + (size_t)nt * 64 * D_,
                       (unsigned)(uintptr_t)(&wlds[0]), 64);
    __syncthreads();

    const bf16* arow = attn + (size_t)(mblk * 128 + wave * 16 + l16) * D_;
    const bf16* l0 = wlds + (size_t)l16 * D_;
    const bf16* l1 = l0 + (size_t)16 * D_;
    const bf16* l2 = l0 + (size_t)32 * D_;
    const bf16* l3 = l0 + (size_t)48 * D_;

    v8f acc[4] = {{}, {}, {}, {}};
    for (int k = 0; k < D_; k += 32) {
        const int ca = k + (hi ? 16 : 0);
        const int kb = k + (hi ? 16 : 0);
        const v16bf A = cat8(ld8(arow + ca), ld8(arow + ca + 8));
        acc[0] = wmma_bf16(A, cat8(ld8(l0 + kb), ld8(l0 + kb + 8)), acc[0]);
        acc[1] = wmma_bf16(A, cat8(ld8(l1 + kb), ld8(l1 + kb + 8)), acc[1]);
        acc[2] = wmma_bf16(A, cat8(ld8(l2 + kb), ld8(l2 + kb + 8)), acc[2]);
        acc[3] = wmma_bf16(A, cat8(ld8(l3 + kb), ld8(l3 + kb + 8)), acc[3]);
    }

    #pragma unroll
    for (int sub = 0; sub < 4; ++sub) {
        const int e = nt * 64 + sub * 16 + l16;
        const float bv = bias[e];
        #pragma unroll
        for (int r = 0; r < 8; ++r) {
            const int grow = mblk * 128 + wave * 16 + r + (hi ? 8 : 0);
            out[(size_t)grow * D_ + e] = acc[sub][r] + bv;
        }
    }
}

// ---------------------------------------------------------------------------
extern "C" void kernel_launch(void* const* d_in, const int* in_sizes, int n_in,
                              void* d_out, int out_size, void* d_ws, size_t ws_size,
                              hipStream_t stream) {
    const float* x     = (const float*)d_in[0];
    const float* w_qkv = (const float*)d_in[1];
    const float* b_qkv = (const float*)d_in[2];
    const float* w_out = (const float*)d_in[3];
    const float* b_out = (const float*)d_in[4];

    char* ws = (char*)d_ws;
    size_t off = 0;
    auto carve = [&](size_t bytes) {
        char* p = ws + off;
        off += (bytes + 255) & ~(size_t)255;
        return p;
    };
    const size_t seg = (size_t)B_ * H_ * N_ * DH_ * sizeof(bf16);
    bf16* xbf    = (bf16*)carve((size_t)BN_ * D_ * sizeof(bf16));
    bf16* wqkvbf = (bf16*)carve((size_t)D3_ * D_ * sizeof(bf16));
    bf16* woutbf = (bf16*)carve((size_t)D_ * D_ * sizeof(bf16));
    bf16* qws    = (bf16*)carve(seg);
    bf16* kws    = (bf16*)carve(seg);
    bf16* vtws   = (bf16*)carve(seg);      // V^T [B,H,Dh,N]
    bf16* attn   = (bf16*)carve((size_t)BN_ * D_ * sizeof(bf16));

    dim3 blk(256);
    const int nx = BN_ * D_ / 8, nwq = D3_ * D_ / 8, nwo = D_ * D_ / 8;
    cvt_kernel<<<(nx  + 255) / 256, blk, 0, stream>>>(x,     xbf,    nx);
    cvt_kernel<<<(nwq + 255) / 256, blk, 0, stream>>>(w_qkv, wqkvbf, nwq);
    cvt_kernel<<<(nwo + 255) / 256, blk, 0, stream>>>(w_out, woutbf, nwo);

    qkv_proj_kernel <<<(BN_ / 128) * (D3_ / 64), blk, 0, stream>>>(
        xbf, wqkvbf, b_qkv, qws, kws, vtws);
    flash_attn_kernel<<<B_ * H_ * (N_ / 128), blk, 0, stream>>>(
        qws, kws, vtws, attn);
    out_proj_kernel <<<(BN_ / 128) * (D_ / 64), blk, 0, stream>>>(
        attn, woutbf, b_out, (float*)d_out);
}
